// Transformer_29137058136850
// MI455X (gfx1250) — matrix operand order repaired and checked by reading, not compiled, not run on hardware
//
#include <hip/hip_runtime.h>
#include <hip/hip_bf16.h>

// ---------------------------------------------------------------------------
// Transformer forward for B=8, S=1024, D=512, H=8 (DH=64) on gfx1250.
// All matmuls run on v_wmma_f32_16x16x32_f16 (f16 operands, f32 accum).
// ---------------------------------------------------------------------------

#define kB 8
#define kS 1024
#define kD 512
#define kH 8
#define kDH 64
#define LN_EPS 1e-3f

typedef _Float16 half16 __attribute__((ext_vector_type(16)));
typedef _Float16 half8  __attribute__((ext_vector_type(8)));
typedef float    float8 __attribute__((ext_vector_type(8)));

union H16 { half16 v; half8 h[2]; };

// ---------------------------------------------------------------------------
// 1) enc_in / dec_in = x + positional embedding  (f32 + f16 copies)
//    even k: sin(pos / 10000^(2k/d)) ; odd k: cos(pos / 10000^(2k/d))
// ---------------------------------------------------------------------------
__global__ __launch_bounds__(256) void add_pos_kernel(
    const float* __restrict__ X, float* __restrict__ Of,
    _Float16* __restrict__ Oh, int n) {
  int i = blockIdx.x * blockDim.x + threadIdx.x;
  if (i >= n) return;
  int k = i % kD;
  int s = (i / kD) % kS;
  // 10000^(2k/d) = exp((2k/d) * ln(10000)); ln(10000)=9.210340371976184
  float angle = (float)s * __expf(-(2.0f * (float)k / (float)kD) * 9.210340371976184f);
  float p = (k & 1) ? __cosf(angle) : __sinf(angle);
  float v = X[i] + p;
  Of[i] = v;
  Oh[i] = (_Float16)v;
}

// ---------------------------------------------------------------------------
// 2) f32 -> f16 conversion (weights)
// ---------------------------------------------------------------------------
__global__ __launch_bounds__(256) void f32_to_f16_kernel(
    const float* __restrict__ X, _Float16* __restrict__ Y, int n) {
  int i = blockIdx.x * blockDim.x + threadIdx.x;
  if (i < n) Y[i] = (_Float16)X[i];
}

// ---------------------------------------------------------------------------
// 3) GEMM: C[M=8192, N=512] = A[M,512](f16) @ W[512,512](f16) (+ R f32)
//    WG = 256 threads = 8 waves (4x2), WG tile 128x64, wave tile 32x32.
//    K-chunk 64: A staged row-major, W staged transposed, 8 WMMAs per chunk.
//    MODE bits: 1 = add residual R, 2 = write f32 Of, 4 = write f16 Oh.
// Fragment layouts (CDNA5 ISA 7.12.2, 16-bit, wave32):
//   A 16x32 : lane m=l&15, half h -> K = (h<8 ? h : h+8) + (l>=16 ? 8 : 0)
//   B 32x16 : lane n=l&15, half h -> K = (h<8 ? h : h+8) + (l>=16 ? 8 : 0)
//   C 16x16 : lane l, vgpr r -> row = r + (l>=16 ? 8 : 0), col = l&15
// ---------------------------------------------------------------------------
template <int MODE>
__global__ __launch_bounds__(256) void gemm_f16_kernel(
    const _Float16* __restrict__ A, const _Float16* __restrict__ W,
    const float* __restrict__ R, float* __restrict__ Of,
    _Float16* __restrict__ Oh) {
  __shared__ __align__(32) _Float16 As[128][64];
  __shared__ __align__(32) _Float16 Wt[64][64];   // transposed W tile [n][k]

  const int t    = threadIdx.x;
  const int lane = t & 31;
  const int wave = t >> 5;
  const int wm   = wave >> 1;            // 0..3
  const int wn   = wave & 1;             // 0..1
  const int lm   = lane & 15;
  const int ofs8  = (lane >= 16) ? 8 : 0;
  const int ofs16 = (lane >= 16) ? 16 : 0;
  const int rofs  = (lane >= 16) ? 8 : 0;
  const int mbase = blockIdx.x * 128;
  const int nbase = blockIdx.y * 64;

  // Per-thread staging assignments (fixed for the whole K loop).
  const int arow = t >> 1, aseg = (t & 1) * 32;
  const _Float16* aptr = A + (size_t)(mbase + arow) * kD + aseg;
  const int wkr = t >> 2, wcc = (t & 3) * 16;
  const _Float16* wptr = W + (size_t)wkr * kD + nbase + wcc;

  float8 acc[2][2] = {};

  for (int kc = 0; kc < kD; kc += 64) {
    // Stage A tile 128x64: one thread loads 32 halves (64B) of one row.
    {
      half16 a0 = *(const half16*)(aptr + kc);
      half16 a1 = *(const half16*)(aptr + kc + 16);
      *(half16*)&As[arow][aseg]      = a0;
      *(half16*)&As[arow][aseg + 16] = a1;
    }
    // Stage W tile 64x64 transposed into Wt[n][k].
    {
      half16 w = *(const half16*)(wptr + (size_t)kc * kD);
#pragma unroll
      for (int j = 0; j < 16; ++j) Wt[wcc + j][wkr] = w[j];
    }
    // Prefetch next K-slab while WMMAs run (global_prefetch_b8).
    if (kc + 64 < kD) {
      __builtin_prefetch(aptr + kc + 64, 0, 0);
      __builtin_prefetch(wptr + (size_t)(kc + 64) * kD, 0, 0);
    }
    __syncthreads();

#pragma unroll
    for (int kk = 0; kk < 64; kk += 32) {
      half16 af[2], bf[2];
#pragma unroll
      for (int i = 0; i < 2; ++i) {
        int row = wm * 32 + i * 16 + lm;
        H16 u;
        u.h[0] = *(const half8*)&As[row][kk + ofs8];
        u.h[1] = *(const half8*)&As[row][kk + 16 + ofs8];
        af[i] = u.v;
      }
#pragma unroll
      for (int j = 0; j < 2; ++j) {
        int col = wn * 32 + j * 16 + lm;
        H16 u;
        u.h[0] = *(const half8*)&Wt[col][kk + ofs8];
        u.h[1] = *(const half8*)&Wt[col][kk + 16 + ofs8];
        bf[j] = u.v;
      }
#pragma unroll
      for (int i = 0; i < 2; ++i)
#pragma unroll
        for (int j = 0; j < 2; ++j)
          acc[i][j] = __builtin_amdgcn_wmma_f32_16x16x32_f16(
              false, af[i], false, bf[j], (short)0, acc[i][j], false, false);
    }
    __syncthreads();
  }

  // Epilogue: compile-time configuration, per-tile base addressing.
#pragma unroll
  for (int i = 0; i < 2; ++i)
#pragma unroll
    for (int j = 0; j < 2; ++j) {
      size_t base = (size_t)(mbase + wm * 32 + i * 16 + rofs) * kD +
                    (size_t)(nbase + wn * 32 + j * 16 + lm);
#pragma unroll
      for (int r = 0; r < 8; ++r) {
        float v = acc[i][j][r];
        size_t idx = base + (size_t)r * kD;
        if (MODE & 1) v += R[idx];
        if (MODE & 2) Of[idx] = v;
        if (MODE & 4) Oh[idx] = (_Float16)v;
      }
    }
}

// ---------------------------------------------------------------------------
// 4) Attention (flash-style, online softmax). scores scaled by 1/H = 0.125.
//    Grid: (S/64, B*H). WG = 128 threads = 4 waves; each wave owns 16 q rows.
//    32-key blocks of K (row-major) and V (transposed) staged in LDS, shared
//    across all 4 waves. MODE bits: 1 = write f32 Of, 2 = write f16 Oh.
// ---------------------------------------------------------------------------
template <int MODE>
__global__ __launch_bounds__(128) void attention_f16_kernel(
    const _Float16* __restrict__ Q, const _Float16* __restrict__ K,
    const _Float16* __restrict__ V, float* __restrict__ Of,
    _Float16* __restrict__ Oh) {
  __shared__ __align__(32) _Float16 Ks[32][64];   // [key][dh]
  __shared__ __align__(32) _Float16 Vt[64][32];   // [dh][key]
  __shared__ __align__(32) float    Ss[4][16][32];
  __shared__ __align__(32) _Float16 Ps[4][16][32];
  __shared__ float mrow[4][16], lrow[4][16], fac[4][16];

  const int t    = threadIdx.x;
  const int lane = t & 31;
  const int wave = t >> 5;
  const int lm   = lane & 15;
  const int ofs8  = (lane >= 16) ? 8 : 0;
  const int ofs16 = (lane >= 16) ? 16 : 0;
  const int rofs  = (lane >= 16) ? 8 : 0;

  const int bh   = blockIdx.y;
  const int b    = bh >> 3;
  const int h    = bh & 7;
  const int hcol = h * kDH;
  const size_t qrowbase = (size_t)b * kS + (size_t)blockIdx.x * 64 + wave * 16;
  const size_t krowbase = (size_t)b * kS;

  // Q fragments for this wave's 16 rows (dh chunks 0..31 and 32..63).
  half16 qa[2];
#pragma unroll
  for (int c = 0; c < 2; ++c) {
    const _Float16* qp = Q + (qrowbase + lm) * kD + hcol + c * 32;
    H16 u;
    u.h[0] = *(const half8*)(qp + ofs8);
    u.h[1] = *(const half8*)(qp + 16 + ofs8);
    qa[c] = u.v;
  }

  float8 o[4] = {};
  if (lane < 16) { mrow[wave][lane] = -1e30f; lrow[wave][lane] = 0.0f; }

  const int ldkey = t >> 2, ldch = (t & 3) * 16;
  const _Float16* kld = K + (krowbase + ldkey) * kD + hcol + ldch;
  const _Float16* vld = V + (krowbase + ldkey) * kD + hcol + ldch;

  for (int kb = 0; kb < kS; kb += 32) {
    __syncthreads();   // previous block's LDS fully consumed
    {
      const size_t gofs = (size_t)kb * kD;
      half16 kv = *(const half16*)(kld + gofs);
      *(half16*)&Ks[ldkey][ldch] = kv;
      half16 vv = *(const half16*)(vld + gofs);
#pragma unroll
      for (int j = 0; j < 16; ++j) Vt[ldch + j][ldkey] = vv[j];
    }
    __syncthreads();

    // Scores: two 16x16 tiles (keys 0..15, 16..31), K-dim = dh = 64.
#pragma unroll
    for (int j = 0; j < 2; ++j) {
      float8 sacc = {};
#pragma unroll
      for (int c = 0; c < 2; ++c) {
        H16 ub;
        ub.h[0] = *(const half8*)&Ks[j * 16 + lm][c * 32 + ofs8];
        ub.h[1] = *(const half8*)&Ks[j * 16 + lm][c * 32 + 16 + ofs8];
        half16 bf = ub.v;
        sacc = __builtin_amdgcn_wmma_f32_16x16x32_f16(
            false, qa[c], false, bf, (short)0, sacc, false, false);
      }
#pragma unroll
      for (int r = 0; r < 8; ++r)
        Ss[wave][r + rofs][j * 16 + lm] = sacc[r] * 0.125f;   // 1/H
    }

    // Online softmax update: lanes 0..15 each own one row.
    if (lane < 16) {
      float mold = mrow[wave][lane];
      float mx = mold;
      float sv[32];
#pragma unroll
      for (int c = 0; c < 32; ++c) { sv[c] = Ss[wave][lane][c]; mx = fmaxf(mx, sv[c]); }
      float rsum = 0.0f;
#pragma unroll
      for (int c = 0; c < 32; ++c) {
        float p = __expf(sv[c] - mx);
        rsum += p;
        Ps[wave][lane][c] = (_Float16)p;
      }
      float f = __expf(mold - mx);
      fac[wave][lane]  = f;
      mrow[wave][lane] = mx;
      lrow[wave][lane] = lrow[wave][lane] * f + rsum;
    }

    // P (16x32 f16) as A-fragment.
    half16 pa;
    {
      H16 u;
      u.h[0] = *(const half8*)&Ps[wave][lm][ofs8];
      u.h[1] = *(const half8*)&Ps[wave][lm][16 + ofs8];
      pa = u.v;
    }
    // Rescale O by exp(m_old - m_new) and accumulate P @ V_block.
#pragma unroll
    for (int tt = 0; tt < 4; ++tt) {
#pragma unroll
      for (int r = 0; r < 8; ++r) o[tt][r] *= fac[wave][r + rofs];
      H16 ub;
      ub.h[0] = *(const half8*)&Vt[tt * 16 + lm][ofs8];
      ub.h[1] = *(const half8*)&Vt[tt * 16 + lm][16 + ofs8];
      half16 bf = ub.v;
      o[tt] = __builtin_amdgcn_wmma_f32_16x16x32_f16(
          false, pa, false, bf, (short)0, o[tt], false, false);
    }
  }

  if (lane < 16) fac[wave][lane] = 1.0f / lrow[wave][lane];

#pragma unroll
  for (int tt = 0; tt < 4; ++tt) {
    size_t base = (qrowbase + rofs) * kD + hcol + tt * 16 + lm;
#pragma unroll
    for (int r = 0; r < 8; ++r) {
      float v = o[tt][r] * fac[wave][r + rofs];
      size_t idx = base + (size_t)r * kD;
      if (MODE & 1) Of[idx] = v;
      if (MODE & 2) Oh[idx] = (_Float16)v;
    }
  }
}

// ---------------------------------------------------------------------------
// 5) out = resid + alpha * (LN(x) * g + b). One wave per row of 512.
//    MODE bits: 1 = write f32 Of, 2 = write f16 Oh.
// ---------------------------------------------------------------------------
template <int MODE>
__global__ __launch_bounds__(256) void ln_add_kernel(
    const float* __restrict__ X, const float* __restrict__ Resid,
    const float* __restrict__ G, const float* __restrict__ Bb, float alpha,
    float* __restrict__ Of, _Float16* __restrict__ Oh) {
  const int lane = threadIdx.x & 31;
  const int wave = threadIdx.x >> 5;
  const size_t row = (size_t)blockIdx.x * 8 + wave;
  const float* x = X + row * kD;

  float v[16];
  float s = 0.0f;
#pragma unroll
  for (int i = 0; i < 16; ++i) { v[i] = x[lane + i * 32]; s += v[i]; }
#pragma unroll
  for (int off = 16; off > 0; off >>= 1) s += __shfl_xor(s, off, 32);
  float mu = s * (1.0f / (float)kD);

  float q = 0.0f;
#pragma unroll
  for (int i = 0; i < 16; ++i) { float d = v[i] - mu; q += d * d; }
#pragma unroll
  for (int off = 16; off > 0; off >>= 1) q += __shfl_xor(q, off, 32);
  float rstd = rsqrtf(q * (1.0f / (float)kD) + LN_EPS);

#pragma unroll
  for (int i = 0; i < 16; ++i) {
    int c = lane + i * 32;
    float ov = Resid[row * kD + c] + alpha * ((v[i] - mu) * rstd * G[c] + Bb[c]);
    if (MODE & 1) Of[row * kD + c] = ov;
    if (MODE & 2) Oh[row * kD + c] = (_Float16)ov;
  }
}

// ---------------------------------------------------------------------------
// Host-side orchestration
// ---------------------------------------------------------------------------
extern "C" void kernel_launch(void* const* d_in, const int* in_sizes, int n_in,
                              void* d_out, int out_size, void* d_ws, size_t ws_size,
                              hipStream_t stream) {
  (void)in_sizes; (void)n_in; (void)out_size; (void)ws_size;

  const float* x_enc = (const float*)d_in[0];
  const float* x_dec = (const float*)d_in[1];
  const float* Wq_e  = (const float*)d_in[2];
  const float* Wk_e  = (const float*)d_in[3];
  const float* Wv_e  = (const float*)d_in[4];
  const float* Wff_e = (const float*)d_in[5];
  const float* g_e   = (const float*)d_in[6];
  const float* b_e   = (const float*)d_in[7];
  const float* Wq_d  = (const float*)d_in[8];
  const float* Wk_d  = (const float*)d_in[9];
  const float* Wv_d  = (const float*)d_in[10];
  const float* Wff_d = (const float*)d_in[11];
  const float* g_d   = (const float*)d_in[12];
  const float* b_d   = (const float*)d_in[13];
  float* out = (float*)d_out;

  const size_t NE    = (size_t)kB * kS * kD;        // 4,194,304 elements
  const size_t F32SZ = NE * sizeof(float);          // 16 MB
  const size_t F16SZ = NE * sizeof(_Float16);       // 8 MB

  char* ws = (char*)d_ws;
  float* F0 = (float*)(ws);                          // enc_in f32 -> i2 f32
  float* F1 = (float*)(ws + F32SZ);                  // dec_in f32
  float* F2 = (float*)(ws + 2 * F32SZ);              // sa_e / sa_d f32
  _Float16* G0 = (_Float16*)(ws + 3 * F32SZ);                 // enc_in_h -> q_d
  _Float16* G1 = (_Float16*)(ws + 3 * F32SZ + 1 * F16SZ);     // dec_in_h
  _Float16* G2 = (_Float16*)(ws + 3 * F32SZ + 2 * F16SZ);     // q_e -> enc_out_h
  _Float16* G3 = (_Float16*)(ws + 3 * F32SZ + 3 * F16SZ);     // k_e -> k_d
  _Float16* G4 = (_Float16*)(ws + 3 * F32SZ + 4 * F16SZ);     // v_e -> v_d
  _Float16* G5 = (_Float16*)(ws + 3 * F32SZ + 5 * F16SZ);     // inter_e_h -> eda_h
  _Float16* WH = (_Float16*)(ws + 3 * F32SZ + 6 * F16SZ);     // 8 half weights

  const dim3 gemm_grid(kB * kS / 128, kD / 64);      // (64, 8)
  const dim3 attn_grid(kS / 64, kB * kH);            // (16, 64)

  // 1) positional embeddings + f16 copies
  add_pos_kernel<<<(NE + 255) / 256, 256, 0, stream>>>(x_enc, F0, G0, (int)NE);
  add_pos_kernel<<<(NE + 255) / 256, 256, 0, stream>>>(x_dec, F1, G1, (int)NE);

  // 2) weights -> f16
  const float* wsrc[8] = {Wq_e, Wk_e, Wv_e, Wff_e, Wq_d, Wk_d, Wv_d, Wff_d};
  const int nw = kD * kD;
  for (int i = 0; i < 8; ++i)
    f32_to_f16_kernel<<<(nw + 255) / 256, 256, 0, stream>>>(
        wsrc[i], WH + (size_t)i * kD * kD, nw);

  // ---- Encoder ----
  gemm_f16_kernel<4><<<gemm_grid, 256, 0, stream>>>(G0, WH + 0 * (size_t)nw, nullptr, nullptr, G2); // q_e
  gemm_f16_kernel<4><<<gemm_grid, 256, 0, stream>>>(G0, WH + 1 * (size_t)nw, nullptr, nullptr, G3); // k_e
  gemm_f16_kernel<4><<<gemm_grid, 256, 0, stream>>>(G0, WH + 2 * (size_t)nw, nullptr, nullptr, G4); // v_e
  attention_f16_kernel<1><<<attn_grid, 128, 0, stream>>>(G2, G3, G4, F2, nullptr);                  // sa_e
  ln_add_kernel<2><<<kB * kS / 8, 256, 0, stream>>>(F2, F0, g_e, b_e, 1.0f, nullptr, G5);           // inter_e
  gemm_f16_kernel<5><<<gemm_grid, 256, 0, stream>>>(G5, WH + 3 * (size_t)nw, F0, nullptr, G2);      // enc_out_h

  // ---- Decoder ----
  gemm_f16_kernel<4><<<gemm_grid, 256, 0, stream>>>(G1, WH + 4 * (size_t)nw, nullptr, nullptr, G0); // q_d
  gemm_f16_kernel<4><<<gemm_grid, 256, 0, stream>>>(G1, WH + 5 * (size_t)nw, nullptr, nullptr, G3); // k_d
  gemm_f16_kernel<4><<<gemm_grid, 256, 0, stream>>>(G1, WH + 6 * (size_t)nw, nullptr, nullptr, G4); // v_d
  attention_f16_kernel<1><<<attn_grid, 128, 0, stream>>>(G0, G3, G4, F2, nullptr);                  // sa_d
  // i2 = dec_in + 2*LN(sa_d)   (reference applies the same LN twice)
  ln_add_kernel<1><<<kB * kS / 8, 256, 0, stream>>>(F2, F1, g_d, b_d, 2.0f, F0, nullptr);
  // eda = attention(enc_out, enc_out, enc_out)
  attention_f16_kernel<2><<<attn_grid, 128, 0, stream>>>(G2, G2, G2, nullptr, G5);
  // dec_out = eda @ Wff_d + i2
  gemm_f16_kernel<3><<<gemm_grid, 256, 0, stream>>>(G5, WH + 7 * (size_t)nw, F0, out, nullptr);
}
